// CQTLayer_21002390077752
// MI455X (gfx1250) — matrix operand order repaired and checked by reading, not compiled, not run on hardware
//
#include <hip/hip_runtime.h>
#include <hip/hip_bf16.h>
#include <math.h>

typedef _Float16 h8  __attribute__((ext_vector_type(8)));
typedef _Float16 h16 __attribute__((ext_vector_type(16)));
typedef float    f8  __attribute__((ext_vector_type(8)));
typedef float    f4  __attribute__((ext_vector_type(4)));

#define HOPSZ   128
#define NBINS   528
#define FRAMES  173
#define NBATCH  2
#define ALEN    ((FRAMES - 1) * HOPSZ)     // 22016 samples per batch
#define MVALID  (NBATCH * FRAMES)          // 346 output rows (b*173 + t)
#define MPAD    384                        // 24 M-tiles of 16 (8 waves x 3 tiles)
#define NTOT    (2 * NBINS)                // 1056 = re rows then im rows
#define SPLITS  16                         // deterministic split-K factor
#define AMIN_F  1e-10f
#define TOPDB   80.0f

// ---------------------------------------------------------------------------
// K1: zero-padded f16 audio + reset global-max slot (ws is poisoned 0xAA).
// ---------------------------------------------------------------------------
__global__ __launch_bounds__(256) void cqt_prep(const float* __restrict__ y,
                                                _Float16* __restrict__ xpad,
                                                unsigned* __restrict__ maxslot,
                                                int xlen, int pad)
{
    int idx = blockIdx.x * 256 + threadIdx.x;
    if (idx == 0) *maxslot = 0u;
    if (idx >= NBATCH * xlen) return;
    int b   = idx / xlen;
    int pos = idx - b * xlen;
    int src = pos - pad;
    float v = 0.0f;
    if (src >= 0 && src < ALEN) v = y[b * ALEN + src];
    xpad[idx] = (_Float16)v;
}

// ---------------------------------------------------------------------------
// K2: split-K WMMA GEMM.  out[m, j] = sum_k xpad[b, t*128+k] * W[j, k]
// Block: 2 N-tiles (32 filter rows) x all 24 M-tiles x one K-split range.
// W chunks are converted fp32->f16 and staged in LDS in B-fragment layout.
// ---------------------------------------------------------------------------
__global__ __launch_bounds__(256) void cqt_gemm(const float* __restrict__ kr,
                                                const float* __restrict__ ki,
                                                const _Float16* __restrict__ xpad,
                                                float* __restrict__ partials,
                                                int L, int nchunks, int xlen)
{
    // 8 fragments = (2 n-tiles x 4 k-chunks) x 32 lanes x 16 halves = 8 KB
    __shared__ _Float16 ldsW[8 * 32 * 16];

    const int tid   = threadIdx.x;
    const int wave  = tid >> 5;
    const int lane  = tid & 31;
    const int j0    = blockIdx.x * 32;   // first filter row of this block
    const int split = blockIdx.y;

    const int cper = (nchunks + SPLITS - 1) / SPLITS;
    int cbeg = split * cper;
    int cend = cbeg + cper; if (cend > nchunks) cend = nchunks;

    // A-fragment row base pointers for the 3 M-tiles this wave owns.
    // A layout (16-bit 16x32, contiguous k as in LDS B): lane<16 -> M=lane, K 0..15;
    // lane>=16 -> M=lane-16, K 16..31.
    const _Float16* abase[3];
#pragma unroll
    for (int a = 0; a < 3; ++a) {
        int m = (wave + 8 * a) * 16 + (lane & 15);
        if (m > MVALID - 1) m = MVALID - 1;      // clamp pad rows (never stored... into output)
        int b = (m >= FRAMES) ? 1 : 0;
        int t = m - b * FRAMES;
        abase[a] = xpad + (size_t)b * xlen + (size_t)t * HOPSZ;
    }
    const int koff8 = (lane >> 4) * 16;

    // Cooperative W loader: thread -> (row wr in 0..31, 16-float k-group wkg in 0..7)
    const int wr  = tid >> 3;
    const int wkg = tid & 7;
    _Float16* ldsDst =
        &ldsW[(((wr >> 4) * 4 + (wkg >> 1)) * 32 + ((wr & 15) + 16 * (wkg & 1))) * 16];
    const int jrow = j0 + wr;
    const float* wsrc = (jrow < NBINS) ? (kr + (size_t)jrow * L)
                                       : (ki + (size_t)(jrow - NBINS) * L);

    f8 acc[2][3];
#pragma unroll
    for (int nt = 0; nt < 2; ++nt)
#pragma unroll
        for (int a = 0; a < 3; ++a)
            acc[nt][a] = (f8){0.f, 0.f, 0.f, 0.f, 0.f, 0.f, 0.f, 0.f};

    for (int cg = cbeg; cg < cend; cg += 4) {
        int g = cend - cg; if (g > 4) g = 4;          // chunks staged this round
        __syncthreads();
        if (wkg < 2 * g) {
            int k0 = cg * 32 + wkg * 16;
            h16 hv;
            if (k0 + 16 <= L) {                        // fast 4x b128 path
                const f4* s4 = (const f4*)(wsrc + k0);
#pragma unroll
                for (int q = 0; q < 4; ++q) {
                    f4 f = s4[q];
#pragma unroll
                    for (int r = 0; r < 4; ++r) hv[q * 4 + r] = (_Float16)f[r];
                }
            } else {                                   // guarded tail (K padding)
#pragma unroll
                for (int i = 0; i < 16; ++i) {
                    int k = k0 + i;
                    hv[i] = (k < L) ? (_Float16)wsrc[k] : (_Float16)0.0f;
                }
            }
            *(h16*)ldsDst = hv;                        // 32B -> 2x ds_store_b128
        }
        __syncthreads();

        for (int c = 0; c < g; ++c) {                  // uniform across block
            int kc = (cg + c) * 32;
            h16 afr[3];
#pragma unroll
            for (int a = 0; a < 3; ++a) {
                const _Float16* p = abase[a] + kc + koff8;
                h8 lo = *(const h8*)p;                 // K = base+0..7
                h8 hi = *(const h8*)(p + 8);           // K = base+8..15
#pragma unroll
                for (int i = 0; i < 8; ++i) { afr[a][i] = lo[i]; afr[a][i + 8] = hi[i]; }
            }
#pragma unroll
            for (int nt = 0; nt < 2; ++nt) {
                h16 bfr = *(const h16*)&ldsW[((nt * 4 + c) * 32 + lane) * 16];
#pragma unroll
                for (int a = 0; a < 3; ++a) {
                    acc[nt][a] = __builtin_amdgcn_wmma_f32_16x16x32_f16(
                        false, afr[a], false, bfr, (short)0, acc[nt][a], false, false);
                }
            }
        }
    }

    // Store partials.  C/D layout: VGPR i -> M = i + 8*(lane>=16), N = lane&15.
#pragma unroll
    for (int nt = 0; nt < 2; ++nt) {
        int jcol = j0 + nt * 16 + (lane & 15);
#pragma unroll
        for (int a = 0; a < 3; ++a) {
            int mrow0 = (wave + 8 * a) * 16 + 8 * (lane >> 4);
            float* dst = partials + ((size_t)split * MPAD + mrow0) * NTOT + jcol;
#pragma unroll
            for (int i = 0; i < 8; ++i) dst[(size_t)i * NTOT] = acc[nt][a][i];
        }
    }
}

// ---------------------------------------------------------------------------
// K3: deterministic split-K reduce -> magnitude, plus order-independent
// global max via atomicMax on float bits (mag >= 0, so u32 order == f32 order).
// ---------------------------------------------------------------------------
__global__ __launch_bounds__(256) void cqt_magmax(const float* __restrict__ partials,
                                                  float* __restrict__ mag,
                                                  unsigned* __restrict__ maxslot)
{
    __shared__ float smax[256];
    int tid = threadIdx.x;
    int idx = blockIdx.x * 256 + tid;
    float mg = 0.0f;
    if (idx < MVALID * NBINS) {
        int m   = idx / NBINS;
        int bin = idx - m * NBINS;
        float re = 0.0f, im = 0.0f;
#pragma unroll
        for (int s = 0; s < SPLITS; ++s) {
            const float* p = partials + ((size_t)s * MPAD + m) * NTOT;
            re += p[bin];
            im += p[bin + NBINS];
        }
        mg = sqrtf(re * re + im * im);
        mag[idx] = mg;                   // [m][bin] == (b, t, k) output layout
    }
    smax[tid] = mg;
    __syncthreads();
    for (int s = 128; s > 0; s >>= 1) {
        if (tid < s) smax[tid] = fmaxf(smax[tid], smax[tid + s]);
        __syncthreads();
    }
    if (tid == 0) atomicMax(maxslot, __float_as_uint(smax[0]));
}

// ---------------------------------------------------------------------------
// K4: dB conversion.  max(log_spec) == 0 by construction, so the TOP_DB
// clamp is simply max(v, -80).
// ---------------------------------------------------------------------------
__global__ __launch_bounds__(256) void cqt_db(const float* __restrict__ mag,
                                              const unsigned* __restrict__ maxslot,
                                              float* __restrict__ out)
{
    int idx = blockIdx.x * 256 + threadIdx.x;
    if (idx >= MVALID * NBINS) return;
    float ref = fmaxf(__uint_as_float(*maxslot), AMIN_F);
    float v = 10.0f * log10f(fmaxf(mag[idx], AMIN_F)) - 10.0f * log10f(ref);
    out[idx] = fmaxf(v, -TOPDB);
}

// ---------------------------------------------------------------------------
extern "C" void kernel_launch(void* const* d_in, const int* in_sizes, int n_in,
                              void* d_out, int out_size, void* d_ws, size_t ws_size,
                              hipStream_t stream)
{
    (void)n_in; (void)out_size; (void)ws_size;
    const float* y  = (const float*)d_in[0];
    const float* kr = (const float*)d_in[1];
    const float* ki = (const float*)d_in[2];
    float* out = (float*)d_out;

    // Derive geometry from input sizes (L = 63864 for the reference config).
    const int L       = in_sizes[1] / NBINS;
    const int pad     = L / 2;
    const int nchunks = (L + 31) / 32;          // K chunks of 32 (zero padded)
    const int xlen    = ALEN + nchunks * 32;    // padded audio length per batch

    // Workspace layout (256B aligned regions).
    size_t xpadBytes = ((size_t)NBATCH * xlen * sizeof(_Float16) + 255) & ~(size_t)255;
    size_t partBytes = (size_t)SPLITS * MPAD * NTOT * sizeof(float);
    size_t magBytes  = ((size_t)MVALID * NBINS * sizeof(float) + 255) & ~(size_t)255;

    char* ws = (char*)d_ws;
    _Float16* xpad    = (_Float16*)ws;
    float*    partials = (float*)(ws + xpadBytes);
    float*    mag      = (float*)(ws + xpadBytes + partBytes);
    unsigned* maxslot  = (unsigned*)(ws + xpadBytes + partBytes + magBytes);

    int prepBlocks = (NBATCH * xlen + 255) / 256;
    cqt_prep<<<prepBlocks, 256, 0, stream>>>(y, xpad, maxslot, xlen, pad);

    dim3 ggrid(NTOT / 32, SPLITS);              // 33 N-groups x 16 K-splits
    cqt_gemm<<<ggrid, 256, 0, stream>>>(kr, ki, xpad, partials, L, nchunks, xlen);

    int elems   = MVALID * NBINS;
    int rblocks = (elems + 255) / 256;
    cqt_magmax<<<rblocks, 256, 0, stream>>>(partials, mag, maxslot);
    cqt_db<<<rblocks, 256, 0, stream>>>(mag, maxslot, out);
}
